// Transformer_24223615550001
// MI455X (gfx1250) — matrix operand order repaired and checked by reading, not compiled, not run on hardware
//
#include <hip/hip_runtime.h>

typedef __attribute__((ext_vector_type(16))) _Float16 v16h;
typedef __attribute__((ext_vector_type(8)))  _Float16 v8h;
typedef __attribute__((ext_vector_type(8)))  float    v8f;

#define DEV __device__ __forceinline__

DEV v16h cat16(v8h lo, v8h hi) {
  return __builtin_shufflevector(lo, hi, 0,1,2,3,4,5,6,7,8,9,10,11,12,13,14,15);
}

DEV v8f wmma16(v16h a, v16h b, v8f c) {
  // D = A(16x32 f16) * B(32x16 f16) + C(16x16 f32)
  return __builtin_amdgcn_wmma_f32_16x16x32_f16(false, a, false, b, (short)0, c, false, false);
}

DEV v8h cvt8(float4 a, float4 b) {
  v8h r;
  r[0]=(_Float16)a.x; r[1]=(_Float16)a.y; r[2]=(_Float16)a.z; r[3]=(_Float16)a.w;
  r[4]=(_Float16)b.x; r[5]=(_Float16)b.y; r[6]=(_Float16)b.z; r[7]=(_Float16)b.w;
  return r;
}

// 16-byte global -> LDS async copy (ASYNCcnt path). Flat shared-pointer low 32
// bits are the LDS byte offset (LDS aperture: addr[63:32]=base, [31:0]=offset).
DEV void async_cp16(_Float16* lds_dst, const _Float16* gsrc) {
  unsigned l = (unsigned)(uintptr_t)lds_dst;
  unsigned long long g = (unsigned long long)(uintptr_t)gsrc;
  asm volatile("global_load_async_to_lds_b128 %0, %1, off"
               :: "v"(l), "v"(g) : "memory");
}

// async loads complete in order per wave: with 4 copies/tile, waiting <=4
// drains the current tile while the prefetched tile's 4 copies stay in flight.
DEV void wait_async_cur()  { asm volatile("s_wait_asynccnt 4" ::: "memory"); }
DEV void wait_async_all()  { asm volatile("s_wait_asynccnt 0" ::: "memory"); }

// ---------------------------------------------------------------------------
// WMMA GEMM on pre-converted f16 operands: C = act(A[M,K] x BT[N,K]^T + bias)
// block = 256 thr = 8 waves; block tile 128x128; wave tile 32x64; K step 32.
// Operand tiles double-buffered via global_load_async_to_lds_b128 so the async
// engine streams tile k+1 while WMMAs consume tile k.
// OUT: 0=f32 [m][n]   1=f16 [m][n]   2=f16 [n][m] (ldc=S, used for V^T)
//      3=f16 head-split [(n>>6)][m][(n&63)] (ldc=S, used for Q/K)
// z-batching: A += z*sAz, C += z*sCz (B shared across z).
// ---------------------------------------------------------------------------
template<int OUT, bool BIAS, bool RELU>
__global__ __launch_bounds__(256)
void gemm_wmma(const _Float16* __restrict__ A, const _Float16* __restrict__ BT,
               void* __restrict__ Cout, const float* __restrict__ bias,
               int M, int N, int K, int lda, int ldb, int ldc,
               long sAz, long sCz)
{
  __shared__ _Float16 sA[2][128][32];   // [buf][m][k]
  __shared__ _Float16 sB[2][128][32];   // [buf][n][k]

  const int z = blockIdx.z;
  A += (long)z * sAz;

  const int t = threadIdx.x, lane = t & 31, w = t >> 5;
  const int hg = lane >> 4, l16 = lane & 15;
  const int wm = w >> 1, wn = w & 1;
  const int m0 = blockIdx.y * 128, n0 = blockIdx.x * 128;

  // staging: 16B chunks; rows r0 (0..63) and r0+64, 4 chunks per 32-half row
  const int r0 = t >> 2, p0 = (t & 3) * 8;

  auto stage = [&](int kk, int bi) {
    async_cp16(&sA[bi][r0     ][p0], A  + (long)(m0 + r0     ) * lda + kk + p0);
    async_cp16(&sA[bi][r0 + 64][p0], A  + (long)(m0 + r0 + 64) * lda + kk + p0);
    async_cp16(&sB[bi][r0     ][p0], BT + (long)(n0 + r0     ) * ldb + kk + p0);
    async_cp16(&sB[bi][r0 + 64][p0], BT + (long)(n0 + r0 + 64) * ldb + kk + p0);
  };

  v8f acc[2][4] = {};
  int buf = 0;
  stage(0, 0);

  for (int kk = 0; kk < K; kk += 32) {
    if (kk + 32 < K) {            // prefetch next tile into the other buffer
      stage(kk + 32, buf ^ 1);    // (its readers finished at prev iter's barrier)
      wait_async_cur();
    } else {
      wait_async_all();
    }
    __syncthreads();

    v16h af[2], bf[4];
    #pragma unroll
    for (int mi = 0; mi < 2; ++mi) {
      const _Float16* p = &sA[buf][wm*32 + mi*16 + l16][0];
      af[mi] = cat16(*(const v8h*)(p + hg*8), *(const v8h*)(p + 16 + hg*8));
    }
    #pragma unroll
    for (int ni = 0; ni < 4; ++ni) {
      const _Float16* p = &sB[buf][wn*64 + ni*16 + l16][0];
      bf[ni] = cat16(*(const v8h*)(p + hg*8), *(const v8h*)(p + 16 + hg*8));
    }
    #pragma unroll
    for (int mi = 0; mi < 2; ++mi)
      #pragma unroll
      for (int ni = 0; ni < 4; ++ni)
        acc[mi][ni] = wmma16(af[mi], bf[ni], acc[mi][ni]);
    __syncthreads();
    buf ^= 1;
  }

  #pragma unroll
  for (int mi = 0; mi < 2; ++mi)
    #pragma unroll
    for (int ni = 0; ni < 4; ++ni)
      #pragma unroll
      for (int v = 0; v < 8; ++v) {
        int m = m0 + wm*32 + mi*16 + v + 8*hg;
        int n = n0 + wn*64 + ni*16 + l16;
        float val = acc[mi][ni][v];
        if (BIAS) val += bias[n];
        if (RELU) val = fmaxf(val, 0.f);
        if (OUT == 0) {
          float* C = (float*)Cout + (long)z * sCz;
          C[(long)m * ldc + n] = val;
        } else if (OUT == 1) {
          _Float16* C = (_Float16*)Cout + (long)z * sCz;
          C[(long)m * ldc + n] = (_Float16)val;
        } else if (OUT == 2) {
          _Float16* C = (_Float16*)Cout + (long)z * sCz;
          C[(long)n * ldc + m] = (_Float16)val;                 // V^T: [n][s]
        } else {
          _Float16* C = (_Float16*)Cout + (long)z * sCz;
          C[((long)(n >> 6) * ldc + m) * 64 + (n & 63)] = (_Float16)val; // [h][s][dk]
        }
      }
}

// ---------------------------------------------------------------------------
// Fused flash attention (a = q.k^T, mask, softmax, a.v), f16 in, f16 out.
// grid (S/64, B*H), block 128 (4 waves). K/V tiles double-buffered via async
// copies, shared by all 4 waves.
// ---------------------------------------------------------------------------
__global__ __launch_bounds__(128)
void flash_attn(const _Float16* __restrict__ q, const _Float16* __restrict__ k,
                const _Float16* __restrict__ vT, const float* __restrict__ mask,
                _Float16* __restrict__ outc)
{
  __shared__ _Float16 sK[2][32][64];   // [buf][t][dk]
  __shared__ _Float16 sV[2][64][32];   // [buf][dv][t]
  __shared__ _Float16 sP[4][16][32];   // per-wave C->A layout bounce
  const int S = 1024;
  const int bh = blockIdx.y;
  const int b = bh >> 3, h = bh & 7;
  const int t = threadIdx.x;
  const int w = t >> 5, lane = t & 31;
  const int hg = lane >> 4, l16 = lane & 15;
  const int q0 = blockIdx.x * 64 + w * 16;

  const _Float16* qp = q + ((size_t)bh * S + q0 + l16) * 64;
  v16h qf[2];
  #pragma unroll
  for (int c = 0; c < 2; ++c)
    qf[c] = cat16(*(const v8h*)(qp + c*32 + hg*8),
                  *(const v8h*)(qp + c*32 + 16 + hg*8));

  v8f acc[4] = {};
  float m8[8], l8[8];
  #pragma unroll
  for (int v = 0; v < 8; ++v) { m8[v] = -3.0e38f; l8[v] = 0.f; }

  const float* mrow = mask + (size_t)b * S;
  const _Float16* kbase = k  + (size_t)bh * S * 64;
  const _Float16* vbase = vT + (size_t)bh * 64 * S;

  const int c1 = t + 128;
  auto stageKV = [&](int tt, int bi) {
    async_cp16(&sK[bi][t  >> 3][(t  & 7)*8], kbase + (size_t)(tt + (t  >> 3))*64 + (t  & 7)*8);
    async_cp16(&sK[bi][c1 >> 3][(c1 & 7)*8], kbase + (size_t)(tt + (c1 >> 3))*64 + (c1 & 7)*8);
    async_cp16(&sV[bi][t  >> 2][(t  & 3)*8], vbase + (size_t)(t  >> 2)*S + tt + (t  & 3)*8);
    async_cp16(&sV[bi][c1 >> 2][(c1 & 3)*8], vbase + (size_t)(c1 >> 2)*S + tt + (c1 & 3)*8);
  };

  int buf = 0;
  stageKV(0, 0);

  for (int tt = 0; tt < S; tt += 32) {
    if (tt + 32 < S) {                     // prefetch next K/V tile
      stageKV(tt + 32, buf ^ 1);
      wait_async_cur();
    } else {
      wait_async_all();
    }
    __syncthreads();

    // ---- scores (16 rows x 32 cols): 4 WMMAs from LDS ----
    v8f sf[2];
    #pragma unroll
    for (int ns = 0; ns < 2; ++ns) {
      const _Float16* kp = &sK[buf][ns*16 + l16][0];
      v16h b0 = cat16(*(const v8h*)(kp + hg*8),      *(const v8h*)(kp + 16 + hg*8));
      v16h b1 = cat16(*(const v8h*)(kp + 32 + hg*8), *(const v8h*)(kp + 48 + hg*8));
      v8f cz = {};
      cz = wmma16(qf[0], b0, cz);
      sf[ns] = wmma16(qf[1], b1, cz);
      float mv = mrow[tt + ns*16 + l16];
      #pragma unroll
      for (int v = 0; v < 8; ++v)
        sf[ns][v] = mv * sf[ns][v] + (1.0f - mv) * (-1e-30f);
    }

    // ---- online softmax over 16-lane row groups ----
    float mt[8];
    #pragma unroll
    for (int v = 0; v < 8; ++v) mt[v] = fmaxf(sf[0][v], sf[1][v]);
    #pragma unroll
    for (int off = 1; off < 16; off <<= 1)
      #pragma unroll
      for (int v = 0; v < 8; ++v)
        mt[v] = fmaxf(mt[v], __shfl_xor(mt[v], off, 32));

    float p0[8], p1[8], rs[8];
    #pragma unroll
    for (int v = 0; v < 8; ++v) {
      float mn = fmaxf(m8[v], mt[v]);
      float sc = __expf(m8[v] - mn);
      p0[v] = __expf(sf[0][v] - mn);
      p1[v] = __expf(sf[1][v] - mn);
      rs[v] = p0[v] + p1[v];
      m8[v] = mn;
      l8[v] *= sc;
      #pragma unroll
      for (int j = 0; j < 4; ++j) acc[j][v] *= sc;
    }
    #pragma unroll
    for (int off = 1; off < 16; off <<= 1)
      #pragma unroll
      for (int v = 0; v < 8; ++v)
        rs[v] += __shfl_xor(rs[v], off, 32);
    #pragma unroll
    for (int v = 0; v < 8; ++v) l8[v] += rs[v];

    // ---- P: C-layout -> A-layout via per-wave LDS patch ----
    #pragma unroll
    for (int v = 0; v < 8; ++v) {
      sP[w][v + 8*hg][l16]      = (_Float16)p0[v];
      sP[w][v + 8*hg][16 + l16] = (_Float16)p1[v];
    }
    asm volatile("s_wait_dscnt 0" ::: "memory");   // same-wave LDS is in-order
    const _Float16* pp = &sP[w][l16][0];
    v16h pA = cat16(*(const v8h*)(pp + hg*8), *(const v8h*)(pp + 16 + hg*8));

    // ---- o += P(16x32) x V(32x64): 4 WMMAs ----
    #pragma unroll
    for (int j = 0; j < 4; ++j) {
      const _Float16* vp = &sV[buf][j*16 + l16][hg*8];
      v16h bv = cat16(*(const v8h*)vp, *(const v8h*)(vp + 16));
      acc[j] = wmma16(pA, bv, acc[j]);
    }
    __syncthreads();
    buf ^= 1;
  }

  _Float16* op = outc + ((size_t)b * S + q0) * 512 + h * 64;
  #pragma unroll
  for (int j = 0; j < 4; ++j)
    #pragma unroll
    for (int v = 0; v < 8; ++v)
      op[(size_t)(v + 8*hg) * 512 + j*16 + l16] = (_Float16)(acc[j][v] / l8[v]);
}

// ---------------------------------------------------------------------------
// out = LayerNorm(a + r)*gamma + beta, dual f32 + f16 output (wave per row)
// ---------------------------------------------------------------------------
__global__ __launch_bounds__(256)
void ln_residual(const float* __restrict__ a, const float* __restrict__ r,
                 float* __restrict__ out, _Float16* __restrict__ out16,
                 const float* __restrict__ gamma, const float* __restrict__ beta)
{
  const int row  = blockIdx.x * 8 + (threadIdx.x >> 5);
  const int lane = threadIdx.x & 31;
  const float* pa = a + (size_t)row * 512;
  const float* pr = r + (size_t)row * 512;
  float vals[16];
  float s = 0.f;
  #pragma unroll
  for (int i = 0; i < 16; ++i) {
    float x = pa[lane + i*32] + pr[lane + i*32];
    vals[i] = x; s += x;
  }
  #pragma unroll
  for (int off = 1; off < 32; off <<= 1) s += __shfl_xor(s, off, 32);
  float mean = s * (1.f/512.f);
  float sq = 0.f;
  #pragma unroll
  for (int i = 0; i < 16; ++i) { float d = vals[i] - mean; sq += d*d; }
  #pragma unroll
  for (int off = 1; off < 32; off <<= 1) sq += __shfl_xor(sq, off, 32);
  float rstd = rsqrtf(sq * (1.f/512.f) + 1e-14f);
  float g = gamma[0], be = beta[0];
  float* po = out + (size_t)row * 512;
  _Float16* ph = out16 + (size_t)row * 512;
  #pragma unroll
  for (int i = 0; i < 16; ++i) {
    float y = (vals[i] - mean) * rstd * g + be;
    po[lane + i*32] = y;
    ph[lane + i*32] = (_Float16)y;
  }
}

// initial: xbuf = x (f32), x16 = (f16)x
__global__ void seed_x(const float4* __restrict__ src, float4* __restrict__ xf,
                       v8h* __restrict__ xh, int n8)
{
  int i = blockIdx.x * blockDim.x + threadIdx.x;
  if (i < n8) {
    float4 a = src[2*i], b = src[2*i + 1];
    xf[2*i] = a; xf[2*i + 1] = b;
    xh[i] = cvt8(a, b);
  }
}

// one-time weight convert + transpose: out[z][n][k] = (f16) in[z][k][n]
__global__ void cvtT(const float* __restrict__ in, _Float16* __restrict__ out,
                     int K, int N)
{
  long base = (long)blockIdx.y * K * N;
  int f = blockIdx.x * 256 + threadIdx.x;
  int kIdx = f / N, n = f % N;
  out[base + (long)n * K + kIdx] = (_Float16)in[base + f];
}

__global__ void copy_f32(const float4* __restrict__ src, float4* __restrict__ dst, int n4)
{
  int i = blockIdx.x * blockDim.x + threadIdx.x;
  if (i < n4) dst[i] = src[i];
}

// ---------------------------------------------------------------------------
extern "C" void kernel_launch(void* const* d_in, const int* in_sizes, int n_in,
                              void* d_out, int out_size, void* d_ws, size_t ws_size,
                              hipStream_t stream)
{
  const float* x     = (const float*)d_in[0];
  const float* mask  = (const float*)d_in[1];
  const float* wq    = (const float*)d_in[2];
  const float* wk    = (const float*)d_in[3];
  const float* wv    = (const float*)d_in[4];
  const float* wo    = (const float*)d_in[5];
  const float* w1    = (const float*)d_in[6];
  const float* b1    = (const float*)d_in[7];
  const float* w2    = (const float*)d_in[8];
  const float* b2    = (const float*)d_in[9];
  const float* gamma = (const float*)d_in[10];
  const float* beta  = (const float*)d_in[11];

  const int B = 8, S = 1024, D = 512, H = 8, DK = 64, DFF = 1024, NB = 2;
  const int M = B * S;

  // workspace (96 MiB total -- fits in the 192 MB L2)
  char* ws = (char*)d_ws;
  float*    xbuf   = (float*)(ws);                     // 16 MiB running x (f32)
  float*    ybuf   = (float*)(ws + (16u << 20));       // 16 MiB sublayer out (f32)
  _Float16* x16    = (_Float16*)(ws + (32u << 20));    //  8 MiB running x (f16)
  _Float16* hid16  = (_Float16*)(ws + (40u << 20));    // 16 MiB FFN hidden (f16)
  _Float16* conc16 = (_Float16*)(ws + (56u << 20));    //  8 MiB attn concat (f16)
  _Float16* q16    = (_Float16*)(ws + (64u << 20));    //  8 MiB [bh][s][64]
  _Float16* k16    = (_Float16*)(ws + (72u << 20));    //  8 MiB [bh][s][64]
  _Float16* vT16   = (_Float16*)(ws + (80u << 20));    //  8 MiB [bh][64][s]
  _Float16* wqT    = (_Float16*)(ws + (88u << 20));    //  1 MiB [i][h*64+k][d]
  _Float16* wkT    = (_Float16*)(ws + (89u << 20));    //  1 MiB
  _Float16* wvT    = (_Float16*)(ws + (90u << 20));    //  1 MiB
  _Float16* woT    = (_Float16*)(ws + (91u << 20));    //  1 MiB [i][n][j]
  _Float16* w1T    = (_Float16*)(ws + (92u << 20));    //  2 MiB [i][n][d]
  _Float16* w2T    = (_Float16*)(ws + (94u << 20));    //  2 MiB [i][n][j]

  // --- one-time: seed x and pre-transpose/convert all weights to f16 ---
  seed_x<<<(M*D/8 + 255)/256, 256, 0, stream>>>((const float4*)x, (float4*)xbuf,
                                                (v8h*)x16, M*D/8);
  cvtT<<<dim3(D*DK/256,  NB*H), 256, 0, stream>>>(wq, wqT, D, DK);
  cvtT<<<dim3(D*DK/256,  NB*H), 256, 0, stream>>>(wk, wkT, D, DK);
  cvtT<<<dim3(D*DK/256,  NB*H), 256, 0, stream>>>(wv, wvT, D, DK);
  cvtT<<<dim3(D*D/256,   NB),   256, 0, stream>>>(wo, woT, H*DK, D);
  cvtT<<<dim3(D*DFF/256, NB),   256, 0, stream>>>(w1, w1T, D, DFF);
  cvtT<<<dim3(DFF*D/256, NB),   256, 0, stream>>>(w2, w2T, DFF, D);

  for (int i = 0; i < NB; ++i) {
    // --- QKV: one GEMM each, z batched over b; head split in epilogue ---
    dim3 gq(512/128, S/128, B);
    gemm_wmma<3,false,false><<<gq, 256, 0, stream>>>(
        x16, wqT + (size_t)i*H*DK*D, q16, nullptr,
        S, H*DK, D, D, D, S, (long)S*D, (long)H*S*DK);
    gemm_wmma<3,false,false><<<gq, 256, 0, stream>>>(
        x16, wkT + (size_t)i*H*DK*D, k16, nullptr,
        S, H*DK, D, D, D, S, (long)S*D, (long)H*S*DK);
    gemm_wmma<2,false,false><<<gq, 256, 0, stream>>>(          // V^T [n][s]
        x16, wvT + (size_t)i*H*DK*D, vT16, nullptr,
        S, H*DK, D, D, D, S, (long)S*D, (long)H*DK*S);

    // --- fused masked attention ---
    flash_attn<<<dim3(S/64, B*H), 128, 0, stream>>>(q16, k16, vT16, mask, conc16);

    // --- output projection ---
    gemm_wmma<0,false,false><<<dim3(D/128, M/128, 1), 256, 0, stream>>>(
        conc16, woT + (size_t)i*H*DK*D, ybuf, nullptr,
        M, D, H*DK, H*DK, H*DK, D, 0, 0);

    ln_residual<<<M/8, 256, 0, stream>>>(xbuf, ybuf, xbuf, x16,
                                         gamma + 2*i, beta + 2*i);

    // --- FFN ---
    gemm_wmma<1,true,true><<<dim3(DFF/128, M/128, 1), 256, 0, stream>>>(
        x16, w1T + (size_t)i*D*DFF, hid16, b1 + (size_t)i*DFF,
        M, DFF, D, D, D, DFF, 0, 0);
    gemm_wmma<0,true,false><<<dim3(D/128, M/128, 1), 256, 0, stream>>>(
        hid16, w2T + (size_t)i*DFF*D, ybuf, b2 + (size_t)i*D,
        M, D, DFF, DFF, DFF, D, 0, 0);

    ln_residual<<<M/8, 256, 0, stream>>>(xbuf, ybuf, xbuf, x16,
                                         gamma + 2*i + 1, beta + 2*i + 1);
  }

  copy_f32<<<(M*D/4 + 255)/256, 256, 0, stream>>>((const float4*)xbuf,
                                                  (float4*)d_out, M*D/4);
}
